// EnhancedSparseAttention_72413148611018
// MI455X (gfx1250) — matrix operand order repaired and checked by reading, not compiled, not run on hardware
//
#include <hip/hip_runtime.h>
#include <hip/hip_bf16.h>
#include <stdint.h>

typedef _Float16 half_t;
typedef __attribute__((ext_vector_type(16))) _Float16 v16h;
typedef __attribute__((ext_vector_type(8)))  _Float16 v8h;
typedef __attribute__((ext_vector_type(8)))  float    v8f;

#define NHEADS 8
#define CCH    256
#define NPIX   4096
#define HD     32
#define NEG_INF (-1e30f)

static __device__ __forceinline__ v8f wmma_f16(v16h a, v16h b, v8f c) {
  // D(16x16 f32) = A(16x32 f16) * B(32x16 f16) + C
  return __builtin_amdgcn_wmma_f32_16x16x32_f16(false, a, false, b, (short)0, c,
                                                false, false);
}

// A-operand halves in this code's contiguous k order (not AMD's table order): K = h + 16*(lane>>4)
// -> 16 contiguous halves at K = 16*hi.
static __device__ __forceinline__ v16h load_a_rowmajor(const half_t* row, int hi) {
  v8h alo = *reinterpret_cast<const v8h*>(row + 16 * hi);
  v8h ahi = *reinterpret_cast<const v8h*>(row + 16 * hi + 8);
  v16h a;
#pragma unroll
  for (int i = 0; i < 8; ++i) { a[i] = alo[i]; a[8 + i] = ahi[i]; }
  return a;
}

// B-operand halves in this code's contiguous k order (matching A): N = lane&15 fixed, K = h + 16*(lane>>4)
// -> 16 contiguous K values when the K-dim is contiguous in memory.
static __device__ __forceinline__ v16h load_b_kcontig(const half_t* p) {
  v8h blo = *reinterpret_cast<const v8h*>(p);
  v8h bhi = *reinterpret_cast<const v8h*>(p + 8);
  v16h b;
#pragma unroll
  for (int i = 0; i < 8; ++i) { b[i] = blo[i]; b[8 + i] = bhi[i]; }
  return b;
}

// ---------------------------------------------------------------------------
// Prep 1: weights f32 -> f16
// ---------------------------------------------------------------------------
__global__ __launch_bounds__(256) void prep_w_kernel(
    const float* __restrict__ Wq, const float* __restrict__ Wk,
    const float* __restrict__ Wv, const float* __restrict__ Wo,
    half_t* __restrict__ W16)  // 4 x 256*256
{
  const int i = blockIdx.x * 256 + threadIdx.x;   // 0..65535
  W16[i]             = (half_t)Wq[i];
  W16[65536 + i]     = (half_t)Wk[i];
  W16[2 * 65536 + i] = (half_t)Wv[i];
  W16[3 * 65536 + i] = (half_t)Wo[i];
}

// ---------------------------------------------------------------------------
// Prep 2: x (C,N) f32 -> xT (N,C) f16 via LDS tile transpose
// ---------------------------------------------------------------------------
__global__ __launch_bounds__(256) void prep_x_kernel(
    const float* __restrict__ x, half_t* __restrict__ xT)
{
  __shared__ float tile[32][33];
  const int c0 = (blockIdx.x & 7) << 5;
  const int n0 = (blockIdx.x >> 3) << 5;
  const int tx = threadIdx.x & 31, ty = threadIdx.x >> 5;
#pragma unroll
  for (int i = 0; i < 4; ++i) {
    const int c = ty + i * 8;
    tile[c][tx] = x[(size_t)(c0 + c) * NPIX + n0 + tx];
  }
  __syncthreads();
#pragma unroll
  for (int i = 0; i < 4; ++i) {
    const int n = ty + i * 8;
    xT[(size_t)(n0 + n) * CCH + c0 + tx] = (half_t)tile[tx][n];
  }
}

// ---------------------------------------------------------------------------
// Stage 1: Q/K/V projections.  A=W16 (row-major f16), B=xT (N,C) f16.
// Q,K -> (N,C); V -> (C,N).
// ---------------------------------------------------------------------------
__global__ __launch_bounds__(128) void qkv_proj_kernel(
    const half_t* __restrict__ W16, const half_t* __restrict__ xT,
    const float* __restrict__ bq, const float* __restrict__ bk,
    const float* __restrict__ bv,
    half_t* __restrict__ Qn, half_t* __restrict__ Kn, half_t* __restrict__ Vc)
{
  const int wave = threadIdx.x >> 5;
  const int lane = threadIdx.x & 31;
  const int lo = lane & 15, hi = lane >> 4;
  const int gw   = blockIdx.x * 4 + wave;   // 0..12287
  const int proj = gw >> 12;                // 0=Q,1=K,2=V
  const int t    = gw & 4095;
  const int m0   = (t >> 8) << 4;
  const int n0   = (t & 255) << 4;

  const half_t* W  = W16 + (size_t)proj * 65536;
  const float*  bb = (proj == 0) ? bq : (proj == 1) ? bk : bv;

  v8f acc = {};
  for (int c0 = 0; c0 < CCH; c0 += 32) {
    v16h a = load_a_rowmajor(W + (size_t)(m0 + lo) * CCH + c0, hi);
    v16h b = load_b_kcontig(xT + (size_t)(n0 + lo) * CCH + c0 + 16 * hi);
    acc = wmma_f16(a, b, acc);
  }

  if (proj == 2) {                          // V -> (C,N)
#pragma unroll
    for (int r = 0; r < 8; ++r) {
      const int o = m0 + 8 * hi + r;
      Vc[(size_t)o * NPIX + n0 + lo] = (half_t)(acc[r] + bb[o]);
    }
  } else {                                  // Q,K -> (N,C): one v8h store
    half_t* dst = (proj == 0) ? Qn : Kn;
    v8h pk;
#pragma unroll
    for (int r = 0; r < 8; ++r) pk[r] = (half_t)(acc[r] + bb[m0 + 8 * hi + r]);
    *reinterpret_cast<v8h*>(dst + (size_t)(n0 + lo) * CCH + m0 + 8 * hi) = pk;
  }
}

// ---------------------------------------------------------------------------
// Stage 2: flash attention; wave = 16 queries of one head; 64-key chunks.
// Per chunk: 4 score WMMAs, base-2 exp softmax with running raw-score max
// (valid upper bound -> identical softmax), P -> LDS, reload as A-operand,
// mask with four 8-byte loads, 2 ones-WMMAs (masked row-sums) + 4 PV WMMAs.
// ---------------------------------------------------------------------------
__global__ __launch_bounds__(128) void attn_kernel(
    const half_t* __restrict__ Qn, const half_t* __restrict__ Kn,
    const half_t* __restrict__ Vc, const unsigned char* __restrict__ mask,
    half_t* __restrict__ attN)
{
  const int wave = threadIdx.x >> 5;
  const int lane = threadIdx.x & 31;
  const int lo = lane & 15, hi = lane >> 4;
  const int tile = blockIdx.x * 4 + wave;   // 0..2047
  const int head = tile >> 8;
  const int q0   = (tile & 255) << 4;
  const float kexp = 0.17677669529663687f * 1.4426950408889634f; // scale*log2(e)

  __shared__ _Float16 pbuf[4][16 * 64];
  _Float16* pb = pbuf[wave];

  const v16h aq = load_a_rowmajor(Qn + (size_t)(q0 + lo) * CCH + head * HD, hi);

  v16h bones;
#pragma unroll
  for (int i = 0; i < 16; ++i) bones[i] = (half_t)1.0f;

  const unsigned char* mH = mask + (size_t)head * NPIX * NPIX;
  const half_t* VcH = Vc + (size_t)head * HD * NPIX;

  float m[8], ssum[8], corr[8];
  v8f o0 = {}, o1 = {};
#pragma unroll
  for (int r = 0; r < 8; ++r) { m[r] = NEG_INF; ssum[r] = 0.f; }

  for (int k0 = 0; k0 < NPIX; k0 += 64) {
    if (k0 + 128 < NPIX)  // mask lookahead: global_prefetch_b8
      __builtin_prefetch(mH + (size_t)(q0 + lo) * NPIX + k0 + 128 + 32 * hi, 0, 1);

    // ---- scores S(16x64) = Q * K^T : four 16x16 tiles ----
    const size_t kb = (size_t)head * HD + 16 * hi;
    v16h bk0 = load_b_kcontig(Kn + (size_t)(k0 + lo) * CCH + kb);
    v16h bk1 = load_b_kcontig(Kn + (size_t)(k0 + 16 + lo) * CCH + kb);
    v16h bk2 = load_b_kcontig(Kn + (size_t)(k0 + 32 + lo) * CCH + kb);
    v16h bk3 = load_b_kcontig(Kn + (size_t)(k0 + 48 + lo) * CCH + kb);
    v8f cz = {};
    v8f s0 = wmma_f16(aq, bk0, cz);
    v8f s1 = wmma_f16(aq, bk1, cz);
    v8f s2 = wmma_f16(aq, bk2, cz);
    v8f s3 = wmma_f16(aq, bk3, cz);

    // ---- online softmax (base-2, scale folded into kexp) ----
#pragma unroll
    for (int r = 0; r < 8; ++r) {
      float rm = fmaxf(fmaxf(s0[r], s1[r]), fmaxf(s2[r], s3[r]));
#pragma unroll
      for (int xm = 1; xm < 16; xm <<= 1) rm = fmaxf(rm, __shfl_xor(rm, xm, 32));
      const float nm  = fmaxf(m[r], rm);
      const float nmk = nm * kexp;
      corr[r] = __builtin_amdgcn_exp2f(fmaf(m[r], kexp, -nmk));
      m[r] = nm;
      const float p0 = __builtin_amdgcn_exp2f(fmaf(s0[r], kexp, -nmk));
      const float p1 = __builtin_amdgcn_exp2f(fmaf(s1[r], kexp, -nmk));
      const float p2 = __builtin_amdgcn_exp2f(fmaf(s2[r], kexp, -nmk));
      const float p3 = __builtin_amdgcn_exp2f(fmaf(s3[r], kexp, -nmk));
      o0[r] *= corr[r];
      o1[r] *= corr[r];
      const int row = (8 * hi + r) * 64;        // C/D layout -> LDS
      pb[row + lo]      = (half_t)p0;
      pb[row + 16 + lo] = (half_t)p1;
      pb[row + 32 + lo] = (half_t)p2;
      pb[row + 48 + lo] = (half_t)p3;
    }

    // ---- reload P as two A-operands; mask via four 8-byte loads ----
    v16h ap0, ap1;
    {
      const _Float16* pr = pb + lo * 64;
      v8h a0 = *reinterpret_cast<const v8h*>(pr + 16 * hi);
      v8h a1 = *reinterpret_cast<const v8h*>(pr + 8 + 16 * hi);
      v8h a2 = *reinterpret_cast<const v8h*>(pr + 32 + 16 * hi);
      v8h a3 = *reinterpret_cast<const v8h*>(pr + 40 + 16 * hi);
#pragma unroll
      for (int i = 0; i < 8; ++i) {
        ap0[i] = a0[i]; ap0[8 + i] = a1[i];
        ap1[i] = a2[i]; ap1[8 + i] = a3[i];
      }
    }
    const unsigned char* mp = mH + (size_t)(q0 + lo) * NPIX + k0 + 16 * hi;
    const unsigned long long w0 = *reinterpret_cast<const unsigned long long*>(mp);
    const unsigned long long w1 = *reinterpret_cast<const unsigned long long*>(mp + 8);
    const unsigned long long w2 = *reinterpret_cast<const unsigned long long*>(mp + 32);
    const unsigned long long w3 = *reinterpret_cast<const unsigned long long*>(mp + 40);
#pragma unroll
    for (int j = 0; j < 8; ++j) {
      if (!((w0 >> (8 * j)) & 1)) ap0[j]     = (half_t)0.0f;
      if (!((w1 >> (8 * j)) & 1)) ap0[8 + j] = (half_t)0.0f;
      if (!((w2 >> (8 * j)) & 1)) ap1[j]     = (half_t)0.0f;
      if (!((w3 >> (8 * j)) & 1)) ap1[8 + j] = (half_t)0.0f;
    }

    // ---- masked row-sums via ones-WMMA + four PV tiles ----
    v8f sumD = wmma_f16(ap1, bones, wmma_f16(ap0, bones, cz));
    v16h bv0a = load_b_kcontig(VcH + (size_t)lo * NPIX + k0 + 16 * hi);
    v16h bv0b = load_b_kcontig(VcH + (size_t)lo * NPIX + k0 + 32 + 16 * hi);
    v16h bv1a = load_b_kcontig(VcH + (size_t)(16 + lo) * NPIX + k0 + 16 * hi);
    v16h bv1b = load_b_kcontig(VcH + (size_t)(16 + lo) * NPIX + k0 + 32 + 16 * hi);
    o0 = wmma_f16(ap0, bv0a, o0);
    o0 = wmma_f16(ap1, bv0b, o0);
    o1 = wmma_f16(ap0, bv1a, o1);
    o1 = wmma_f16(ap1, bv1b, o1);
#pragma unroll
    for (int r = 0; r < 8; ++r) ssum[r] = fmaf(ssum[r], corr[r], sumD[r]);
  }

  // ---- normalize, store att in (N,C) for the O-projection B-operand ----
#pragma unroll
  for (int r = 0; r < 8; ++r) {
    const float inv = __builtin_amdgcn_rcpf(ssum[r]);
    const size_t row = (size_t)(q0 + 8 * hi + r) * CCH + head * HD;
    attN[row + lo]      = (half_t)(o0[r] * inv);
    attN[row + 16 + lo] = (half_t)(o1[r] * inv);
  }
}

// ---------------------------------------------------------------------------
// Stage 3a: output projection + bias + residual -> z (C,N) f32
// ---------------------------------------------------------------------------
__global__ __launch_bounds__(128) void out_proj_kernel(
    const half_t* __restrict__ attN, const half_t* __restrict__ W16,
    const float* __restrict__ bo, const float* __restrict__ x,
    float* __restrict__ z)
{
  const int wave = threadIdx.x >> 5;
  const int lane = threadIdx.x & 31;
  const int lo = lane & 15, hi = lane >> 4;
  const int t  = blockIdx.x * 4 + wave;     // 0..4095
  const int m0 = (t >> 8) << 4;
  const int n0 = (t & 255) << 4;
  const half_t* Wo16 = W16 + (size_t)3 * 65536;

  v8f acc = {};
  for (int c0 = 0; c0 < CCH; c0 += 32) {
    v16h a = load_a_rowmajor(Wo16 + (size_t)(m0 + lo) * CCH + c0, hi);
    v16h b = load_b_kcontig(attN + (size_t)(n0 + lo) * CCH + c0 + 16 * hi);
    acc = wmma_f16(a, b, acc);
  }
#pragma unroll
  for (int r = 0; r < 8; ++r) {
    const int o = m0 + 8 * hi + r;
    const size_t idx = (size_t)o * NPIX + n0 + lo;
    z[idx] = acc[r] + bo[o] + x[idx];
  }
}

// ---------------------------------------------------------------------------
// Stage 3b: channel LayerNorm per pixel
// ---------------------------------------------------------------------------
__global__ __launch_bounds__(256) void layernorm_kernel(
    const float* __restrict__ z, const float* __restrict__ gamma,
    const float* __restrict__ beta, float* __restrict__ out)
{
  const int tx = threadIdx.x & 31;
  const int ty = threadIdx.x >> 5;
  const int n  = blockIdx.x * 32 + tx;

  float s = 0.f, sq = 0.f;
#pragma unroll
  for (int i = 0; i < 32; ++i) {
    const float v = z[(size_t)(ty + i * 8) * NPIX + n];
    s += v; sq += v * v;
  }
  __shared__ float sh_s[8][32], sh_q[8][32];
  __shared__ float sh_mu[32], sh_rs[32];
  sh_s[ty][tx] = s; sh_q[ty][tx] = sq;
  __syncthreads();
  if (ty == 0) {
    float ts = 0.f, tq = 0.f;
#pragma unroll
    for (int j = 0; j < 8; ++j) { ts += sh_s[j][tx]; tq += sh_q[j][tx]; }
    const float mu  = ts * (1.0f / 256.0f);
    const float var = tq * (1.0f / 256.0f) - mu * mu;
    sh_mu[tx] = mu;
    sh_rs[tx] = rsqrtf(var + 1e-5f);
  }
  __syncthreads();
  const float mu = sh_mu[tx], rs = sh_rs[tx];
#pragma unroll
  for (int i = 0; i < 32; ++i) {
    const int c = ty + i * 8;
    const float v = z[(size_t)c * NPIX + n];
    out[(size_t)c * NPIX + n] = (v - mu) * rs * gamma[c] + beta[c];
  }
}

// ---------------------------------------------------------------------------
extern "C" void kernel_launch(void* const* d_in, const int* in_sizes, int n_in,
                              void* d_out, int out_size, void* d_ws, size_t ws_size,
                              hipStream_t stream) {
  (void)in_sizes; (void)n_in; (void)out_size; (void)ws_size;
  const float*         x    = (const float*)d_in[0];
  const unsigned char* mask = (const unsigned char*)d_in[1];  // jnp bool = 1 byte
  const float* Wq = (const float*)d_in[2];
  const float* bq = (const float*)d_in[3];
  const float* Wk = (const float*)d_in[4];
  const float* bk = (const float*)d_in[5];
  const float* Wv = (const float*)d_in[6];
  const float* bv = (const float*)d_in[7];
  const float* Wo = (const float*)d_in[8];
  const float* bo = (const float*)d_in[9];
  const float* gamma = (const float*)d_in[10];
  const float* beta  = (const float*)d_in[11];
  float* out = (float*)d_out;

  // workspace: W16(512KB) | xT(2MB) | Qn(2MB) | Kn(2MB) | Vc(2MB) | attN(2MB) | z(4MB)
  half_t* W16  = (half_t*)d_ws;
  half_t* xT   = W16  + (size_t)4 * 65536;
  half_t* Qn   = xT   + (size_t)NPIX * CCH;
  half_t* Kn   = Qn   + (size_t)NPIX * CCH;
  half_t* Vc   = Kn   + (size_t)NPIX * CCH;
  half_t* attN = Vc   + (size_t)NPIX * CCH;
  float*  z    = (float*)(attN + (size_t)NPIX * CCH);

  prep_w_kernel<<<256, 256, 0, stream>>>(Wq, Wk, Wv, Wo, W16);
  prep_x_kernel<<<1024, 256, 0, stream>>>(x, xT);
  qkv_proj_kernel<<<3072, 128, 0, stream>>>(W16, xT, bq, bk, bv, Qn, Kn, Vc);
  attn_kernel<<<512, 128, 0, stream>>>(Qn, Kn, Vc, mask, attN);
  out_proj_kernel<<<1024, 128, 0, stream>>>(attN, W16, bo, x, z);
  layernorm_kernel<<<128, 256, 0, stream>>>(z, gamma, beta, out);
}
